// GCNNet_4672924418108
// MI455X (gfx1250) — compile-verified, matrix operand order checked by reading, not run on hardware
//
#include <hip/hip_runtime.h>
#include <hip/hip_bf16.h>
#include <math.h>

typedef __attribute__((ext_vector_type(16))) _Float16 v16h;
typedef __attribute__((ext_vector_type(8)))  _Float16 v8h;
typedef __attribute__((ext_vector_type(8)))  float    v8f;

#define DH 128   // hidden dim (both layers)
#define GGR 128  // num graphs

// ---------------------------------------------------------------- helpers
__device__ __forceinline__ float gelu_exact(float x) {
  return 0.5f * x * (1.0f + erff(x * 0.70710678118654752f));
}

// float atomic-max via sign-split int/uint atomics (init must be -inf)
__device__ __forceinline__ void atomicMaxF32(float* addr, float v) {
  if (v >= 0.0f) atomicMax((int*)addr, __float_as_int(v));
  else           atomicMin((unsigned int*)addr, __float_as_uint(v));
}

__device__ __forceinline__ v16h cat8(v8h lo, v8h hi) {
  v16h r;
#pragma unroll
  for (int i = 0; i < 8; ++i) { r[i] = lo[i]; r[i + 8] = hi[i]; }
  return r;
}

__global__ void fill_f32(float* p, float v, int count) {
  int i = blockIdx.x * blockDim.x + threadIdx.x;
  if (i < count) p[i] = v;
}

// ---------------------------------------------------------------- operand packing
// Ah[row][kk] = (kk < K) ? (f16)A[row][kk] : 0 ; row-major, padded to Kpad
__global__ void pack_a_f16(const float* __restrict__ A, _Float16* __restrict__ Ah,
                           int M, int K, int Kpad) {
  int i = blockIdx.x * blockDim.x + threadIdx.x;
  if (i >= M * Kpad) return;
  int row = i / Kpad, kk = i - row * Kpad;
  Ah[i] = (kk < K) ? (_Float16)A[(size_t)row * K + kk] : (_Float16)0.0f;
}

// Bt[c][kk] = (kk < K) ? (f16)W[kk][c] : 0 ; W is [K][128] row-major
__global__ void pack_bt_f16(const float* __restrict__ W, _Float16* __restrict__ Bt,
                            int K, int Kpad) {
  int i = blockIdx.x * blockDim.x + threadIdx.x;
  if (i >= DH * Kpad) return;
  int c = i / Kpad, kk = i - c * Kpad;
  Bt[i] = (kk < K) ? (_Float16)W[(size_t)kk * DH + c] : (_Float16)0.0f;
}

// ---------------------------------------------------------------- WMMA GEMM
// C[M x 128] = Ah[M x Kpad](f16) * Bt^T (Bt is [128][Kpad] f16), f32 accumulate.
// One wave computes a full 16x128 row stripe: 8 accumulators, A loaded once per
// k-step, all fragment loads are contiguous global_load_b128 (no predication).
// Fragment layouts per CDNA5 ISA 7.12.2 (wave32).
__global__ void __launch_bounds__(256)
gemm_f16wmma(const _Float16* __restrict__ Ah, const _Float16* __restrict__ Bt,
             float* __restrict__ C, int M, int Kpad) {
  const int lane    = threadIdx.x & 31;
  const int wave    = threadIdx.x >> 5;
  const int half_id = lane >> 4;             // 0: lanes 0-15, 1: lanes 16-31
  const int lp      = lane & 15;
  const int rowBase = (blockIdx.x * (blockDim.x >> 5) + wave) * 16;
  if (rowBase >= M) return;                  // wave-uniform exit (EXEC stays full)

  int arow = rowBase + lp;
  if (arow >= M) arow = M - 1;               // safety clamp (M%16==0 normally)
  const _Float16* Arow = Ah + (size_t)arow * Kpad;

  v8f acc[8];
#pragma unroll
  for (int t = 0; t < 8; ++t) acc[t] = (v8f){};

  for (int kb = 0; kb < Kpad; kb += 32) {
    // lane-group K runs: {k0..k0+7} and {k0+16..k0+23}, k0 = kb + 8*half_id
    const int k0 = kb + 8 * half_id;
    const int k1 = k0 + 16;
    const v16h a = cat8(*(const v8h*)(Arow + k0), *(const v8h*)(Arow + k1));
#pragma unroll
    for (int t = 0; t < 8; ++t) {
      const _Float16* Bcol = Bt + (size_t)(t * 16 + lp) * Kpad;
      const v16h b = cat8(*(const v8h*)(Bcol + k0), *(const v8h*)(Bcol + k1));
      acc[t] = __builtin_amdgcn_wmma_f32_16x16x32_f16(
          false, a, false, b, (short)0, acc[t], false, false);
    }
  }

  // D layout: VGPR r -> row (rowBase + r + 8*half_id), col (t*16 + lp)
#pragma unroll
  for (int t = 0; t < 8; ++t) {
#pragma unroll
    for (int r = 0; r < 8; ++r) {
      const int orow = rowBase + r + 8 * half_id;
      if (orow < M) C[(size_t)orow * DH + t * 16 + lp] = acc[t][r];
    }
  }
}

// ---------------------------------------------------------------- attention dots
// a_s[i] = dot(H[i,:], att_s), a_d[i] = dot(H[i,:], att_d). One wave per node.
__global__ void attn_dots(const float* __restrict__ H,
                          const float* __restrict__ att_s,
                          const float* __restrict__ att_d,
                          float* __restrict__ a_s, float* __restrict__ a_d, int n) {
  const int node = blockIdx.x * (blockDim.x >> 5) + (threadIdx.x >> 5);
  if (node >= n) return;
  const int lane = threadIdx.x & 31;
  const float4 hv = ((const float4*)(H + (size_t)node * DH))[lane];
  const float4 sv = ((const float4*)att_s)[lane];
  const float4 dv = ((const float4*)att_d)[lane];
  float ss = hv.x * sv.x + hv.y * sv.y + hv.z * sv.z + hv.w * sv.w;
  float dd = hv.x * dv.x + hv.y * dv.y + hv.z * dv.z + hv.w * dv.w;
#pragma unroll
  for (int off = 16; off; off >>= 1) {
    ss += __shfl_xor(ss, off, 32);
    dd += __shfl_xor(dd, off, 32);
  }
  if (lane == 0) { a_s[node] = ss; a_d[node] = dd; }
}

// ---------------------------------------------------------------- edge passes
__device__ __forceinline__ void edge_sd(const int* ei, int e, int E, int& s, int& d) {
  if (e < E) { s = ei[e]; d = ei[E + e]; }
  else       { s = e - E; d = e - E; }     // self-loops appended
}

__global__ void edge_logits_max(const int* __restrict__ ei,
                                const float* __restrict__ a_s,
                                const float* __restrict__ a_d,
                                float* __restrict__ e_buf,
                                float* __restrict__ mmax, int E, int n) {
  const int e = blockIdx.x * blockDim.x + threadIdx.x;
  if (e >= E + n) return;
  int s, d; edge_sd(ei, e, E, s, d);
  float v = a_s[s] + a_d[d];
  v = (v > 0.0f) ? v : 0.2f * v;           // leaky_relu(0.2)
  e_buf[e] = v;
  atomicMaxF32(&mmax[d], v);
}

__global__ void edge_expsum(const int* __restrict__ ei,
                            float* __restrict__ e_buf,
                            const float* __restrict__ mmax,
                            float* __restrict__ denom, int E, int n) {
  const int e = blockIdx.x * blockDim.x + threadIdx.x;
  if (e >= E + n) return;
  int s, d; edge_sd(ei, e, E, s, d);
  const float ex = expf(e_buf[e] - mmax[d]);
  e_buf[e] = ex;                            // overwrite in place
  atomicAdd(&denom[d], ex);
}

// One wave per edge: AGG[dst,:] += alpha * H[src,:]   (128 floats = 32 x float4)
__global__ void edge_aggregate(const int* __restrict__ ei,
                               const float* __restrict__ e_buf,
                               const float* __restrict__ denom,
                               const float* __restrict__ H,
                               float* __restrict__ AGG, int E, int n) {
  const int e = blockIdx.x * (blockDim.x >> 5) + (threadIdx.x >> 5);
  if (e >= E + n) return;
  const int lane = threadIdx.x & 31;
  int s, d; edge_sd(ei, e, E, s, d);
  const float alpha = e_buf[e] / denom[d];
  const float4 hv = ((const float4*)(H + (size_t)s * DH))[lane];
  float* out = AGG + (size_t)d * DH + lane * 4;
  atomicAdd(out + 0, alpha * hv.x);
  atomicAdd(out + 1, alpha * hv.y);
  atomicAdd(out + 2, alpha * hv.z);
  atomicAdd(out + 3, alpha * hv.w);
}

// out = (apply_gelu ? gelu(in + bias) : in + bias)
__global__ void bias_act(const float* __restrict__ in, const float* __restrict__ bias,
                         float* __restrict__ out, int n, int apply_gelu) {
  const int i = blockIdx.x * blockDim.x + threadIdx.x;
  if (i >= n * DH) return;
  float v = in[i] + bias[i & (DH - 1)];
  if (apply_gelu) v = gelu_exact(v);
  out[i] = v;
}

// ---------------------------------------------------------------- pooling + head
__global__ void pool_sum(const float* __restrict__ H, const int* __restrict__ batch,
                         const unsigned char* __restrict__ pos,
                         float* __restrict__ sums, float* __restrict__ cnt, int n) {
  const int node = blockIdx.x * (blockDim.x >> 5) + (threadIdx.x >> 5);
  if (node >= n) return;
  if (!pos[node]) return;                   // uniform across the wave
  const int lane = threadIdx.x & 31;
  const int g = batch[node];
  const float4 hv = ((const float4*)(H + (size_t)node * DH))[lane];
  float* out = sums + (size_t)g * DH + lane * 4;
  atomicAdd(out + 0, hv.x);
  atomicAdd(out + 1, hv.y);
  atomicAdd(out + 2, hv.z);
  atomicAdd(out + 3, hv.w);
  if (lane == 0) atomicAdd(&cnt[g], 1.0f);
}

// logits[g,:] = sums[g,:]/max(cnt,1);  scores[g] = gelu(logits[g,:]) @ Wfc + bfc
__global__ void pool_finish(const float* __restrict__ sums, const float* __restrict__ cnt,
                            const float* __restrict__ Wfc, const float* __restrict__ bfc,
                            float* __restrict__ scores, float* __restrict__ logits) {
  const int g = blockIdx.x * (blockDim.x >> 5) + (threadIdx.x >> 5);
  if (g >= GGR) return;
  const int lane = threadIdx.x & 31;
  const float c = fmaxf(cnt[g], 1.0f);
  float partial = 0.0f;
#pragma unroll
  for (int j = 0; j < 4; ++j) {
    const int f = lane * 4 + j;
    const float lg = sums[g * DH + f] / c;
    logits[g * DH + f] = lg;
    partial += gelu_exact(lg) * Wfc[f];
  }
#pragma unroll
  for (int off = 16; off; off >>= 1) partial += __shfl_xor(partial, off, 32);
  if (lane == 0) scores[g] = partial + bfc[0];
}

// ---------------------------------------------------------------- launch
extern "C" void kernel_launch(void* const* d_in, const int* in_sizes, int n_in,
                              void* d_out, int out_size, void* d_ws, size_t ws_size,
                              hipStream_t stream) {
  (void)n_in; (void)out_size; (void)ws_size;
  const float*         x     = (const float*)d_in[0];
  const int*           ei    = (const int*)d_in[1];
  const int*           batch = (const int*)d_in[2];
  const unsigned char* pos   = (const unsigned char*)d_in[3];
  const float* W1  = (const float*)d_in[5];
  const float* as1 = (const float*)d_in[6];
  const float* ad1 = (const float*)d_in[7];
  const float* b1  = (const float*)d_in[8];
  const float* W2  = (const float*)d_in[9];
  const float* as2 = (const float*)d_in[10];
  const float* ad2 = (const float*)d_in[11];
  const float* b2  = (const float*)d_in[12];
  const float* Wfc = (const float*)d_in[13];
  const float* bfc = (const float*)d_in[14];

  const int n     = in_sizes[2];        // 50000 nodes
  const int E     = in_sizes[1] / 2;    // 800000 edges
  const int Din   = in_sizes[0] / n;    // 300
  const int tot   = E + n;              // edges + self-loops
  const int Kpad1 = (Din + 31) & ~31;   // 320
  const int Kpad2 = DH;                 // 128 (already multiple of 32)

  // workspace layout
  float* buf0  = (float*)d_ws;                   // [n,128]
  float* buf1  = buf0  + (size_t)n * DH;         // [n,128]
  float* a_s   = buf1  + (size_t)n * DH;         // [n]
  float* a_d   = a_s   + n;                      // [n]
  float* mmax  = a_d   + n;                      // [n]
  float* denom = mmax  + n;                      // [n]
  float* e_buf = denom + n;                      // [E+n]
  float* sums  = e_buf + tot;                    // [128,128]
  float* cnt   = sums  + (size_t)GGR * DH;       // [128]
  // f16 packed operands (16B-aligned)
  size_t fofs = (size_t)(cnt + GGR - (float*)d_ws);
  fofs = (fofs + 3) & ~(size_t)3;
  _Float16* Ah = (_Float16*)((float*)d_ws + fofs);     // [n, Kpad1] max
  _Float16* Bt = Ah + (size_t)n * Kpad1;               // [128, Kpad1] max

  float* scores = (float*)d_out;                 // [128]
  float* logits = scores + GGR;                  // [128,128]

  const dim3 blk(256);
  const int gemmB  = (n + 127) / 128;   // 8 waves/block, 16 rows/wave
  const int nWaveB = (n + 7) / 8;       // 1 wave per node
  const int eWaveB = (tot + 7) / 8;     // 1 wave per edge
  const int nElB   = (n * DH + 255) / 256;
  const int nB     = (n + 255) / 256;
  const int eB     = (tot + 255) / 256;

  // ---------------- layer 1 ----------------
  pack_a_f16     <<<(n * Kpad1 + 255) / 256, blk, 0, stream>>>(x, Ah, n, Din, Kpad1);
  pack_bt_f16    <<<(DH * Kpad1 + 255) / 256, blk, 0, stream>>>(W1, Bt, Din, Kpad1);
  gemm_f16wmma   <<<gemmB, blk, 0, stream>>>(Ah, Bt, buf0, n, Kpad1);
  attn_dots      <<<nWaveB, blk, 0, stream>>>(buf0, as1, ad1, a_s, a_d, n);
  fill_f32       <<<nElB, blk, 0, stream>>>(buf1, 0.0f, n * DH);
  fill_f32       <<<nB,   blk, 0, stream>>>(mmax, -INFINITY, n);
  fill_f32       <<<nB,   blk, 0, stream>>>(denom, 0.0f, n);
  edge_logits_max<<<eB,   blk, 0, stream>>>(ei, a_s, a_d, e_buf, mmax, E, n);
  edge_expsum    <<<eB,   blk, 0, stream>>>(ei, e_buf, mmax, denom, E, n);
  edge_aggregate <<<eWaveB, blk, 0, stream>>>(ei, e_buf, denom, buf0, buf1, E, n);
  bias_act       <<<nElB, blk, 0, stream>>>(buf1, b1, buf0, n, 1);   // gelu

  // ---------------- layer 2 ----------------
  pack_a_f16     <<<(n * Kpad2 + 255) / 256, blk, 0, stream>>>(buf0, Ah, n, DH, Kpad2);
  pack_bt_f16    <<<(DH * Kpad2 + 255) / 256, blk, 0, stream>>>(W2, Bt, DH, Kpad2);
  gemm_f16wmma   <<<gemmB, blk, 0, stream>>>(Ah, Bt, buf1, n, Kpad2);
  attn_dots      <<<nWaveB, blk, 0, stream>>>(buf1, as2, ad2, a_s, a_d, n);
  fill_f32       <<<nElB, blk, 0, stream>>>(buf0, 0.0f, n * DH);
  fill_f32       <<<nB,   blk, 0, stream>>>(mmax, -INFINITY, n);
  fill_f32       <<<nB,   blk, 0, stream>>>(denom, 0.0f, n);
  edge_logits_max<<<eB,   blk, 0, stream>>>(ei, a_s, a_d, e_buf, mmax, E, n);
  edge_expsum    <<<eB,   blk, 0, stream>>>(ei, e_buf, mmax, denom, E, n);
  edge_aggregate <<<eWaveB, blk, 0, stream>>>(ei, e_buf, denom, buf1, buf0, E, n);
  bias_act       <<<nElB, blk, 0, stream>>>(buf0, b2, buf1, n, 0);   // no gelu

  // ---------------- pooling + head ----------------
  fill_f32   <<<(GGR * DH + 255) / 256, blk, 0, stream>>>(sums, 0.0f, GGR * DH);
  fill_f32   <<<1, blk, 0, stream>>>(cnt, 0.0f, GGR);
  pool_sum   <<<nWaveB, blk, 0, stream>>>(buf1, batch, pos, sums, cnt, n);
  pool_finish<<<(GGR + 7) / 8, blk, 0, stream>>>(sums, cnt, Wfc, bfc, scores, logits);
}
